// TransformerBlock_39152921870720
// MI455X (gfx1250) — matrix operand order repaired and checked by reading, not compiled, not run on hardware
//
#include <hip/hip_runtime.h>
#include <hip/hip_bf16.h>

// ---------------------------------------------------------------------------
// Transformer block for MI455X (gfx1250, wave32, WMMA).
// bf16 v_wmma_f32_16x16x32_bf16 for all matmuls, fp32 accumulation,
// fp32 LayerNorm/residuals. Global->LDS staging uses CDNA5 async-to-LDS
// (ASYNCcnt) where the layout is copy-through; softmax uses a wave32
// row-ownership scheme (1 q-row per lane) instead of cross-lane permutes.
// ---------------------------------------------------------------------------

typedef __attribute__((ext_vector_type(16))) __bf16 v16bf;
typedef __attribute__((ext_vector_type(8)))  float  v8f;

union BFrag {
    v16bf v;
    uint4 q[2];
    unsigned u[8];
};

__device__ __forceinline__ v8f wmma_bf16(const v16bf& a, const v16bf& b, const v8f& c) {
    return __builtin_amdgcn_wmma_f32_16x16x32_bf16(false, a, false, b, (short)0, c, false, false);
}

// CDNA5 async global->LDS copy (tracked by ASYNCcnt, no VGPR data path).
// dsaddr VGPR = LDS byte address (low 32 bits of the flat pointer).
__device__ __forceinline__ void async_ld_b128(void* lds, const void* gsrc) {
    unsigned l32 = (unsigned)(unsigned long long)lds;
    asm volatile("global_load_async_to_lds_b128 %0, %1, off"
                 :: "v"(l32), "v"(gsrc) : "memory");
}
__device__ __forceinline__ void wait_async0() {
    asm volatile("s_wait_asynccnt 0" ::: "memory");
}
// compiler-only fence: same-wave LDS is executed in-order by hardware (DScnt).
__device__ __forceinline__ void lds_fence() {
    asm volatile("" ::: "memory");
}

#define LNE 1e-5f
#define S_LEN 4096
#define NHEAD 8
#define DHEAD 32
#define DMODEL 256

// ---------------------------------------------------------------------------
__global__ void cast_f32_bf16(const float* __restrict__ in, __bf16* __restrict__ out, int n) {
    int i = blockIdx.x * blockDim.x + threadIdx.x;
    if (i < n) out[i] = (__bf16)in[i];
}

// ---------------------------------------------------------------------------
// LayerNorm over D=256, one row per 256-thread block, bf16 output.
// ---------------------------------------------------------------------------
__global__ __launch_bounds__(256)
void layernorm_bf16(const float* __restrict__ x, const float* __restrict__ g,
                    const float* __restrict__ bta, __bf16* __restrict__ out) {
    int row = blockIdx.x;
    float v = x[(size_t)row * DMODEL + threadIdx.x];
    float s1 = v, s2 = v * v;
    for (int st = 16; st >= 1; st >>= 1) {
        s1 += __shfl_xor(s1, st, 32);
        s2 += __shfl_xor(s2, st, 32);
    }
    __shared__ float sh1[8], sh2[8];
    int w = threadIdx.x >> 5, lane = threadIdx.x & 31;
    if (lane == 0) { sh1[w] = s1; sh2[w] = s2; }
    __syncthreads();
    float t1 = 0.f, t2 = 0.f;
    #pragma unroll
    for (int i = 0; i < 8; ++i) { t1 += sh1[i]; t2 += sh2[i]; }
    float mu  = t1 * (1.f / DMODEL);
    float var = t2 * (1.f / DMODEL) - mu * mu;
    float r   = rsqrtf(var + LNE);
    out[(size_t)row * DMODEL + threadIdx.x] =
        (__bf16)((v - mu) * r * g[threadIdx.x] + bta[threadIdx.x]);
}

// ---------------------------------------------------------------------------
// GEMM-NT:  C[M,N] = A[M,K] * Bw[N,K]^T  (bf16 inputs, f32 accum)
// 256 thr = 8 waves (4x2), block tile 128x64, wave tile 32x32, K-step 32.
// Tiles staged with async-to-LDS (row stride 40 bf16 => 80B, 16B aligned).
// ---------------------------------------------------------------------------
template <int EPI>
__global__ __launch_bounds__(256)
void gemm_nt(const __bf16* __restrict__ A, const __bf16* __restrict__ Bw,
             int M, int N, int K,
             const float* __restrict__ bias, const float* __restrict__ resid,
             void* __restrict__ outp, int ropeFlag) {
    __shared__ __bf16 As[128 * 40];
    __shared__ __bf16 Bs[64 * 40];

    const int t    = threadIdx.x;
    const int lane = t & 31;
    const int w    = t >> 5;
    const int wm   = w >> 1;
    const int wn   = w & 1;
    const int lhi  = lane >> 4;
    const int llo  = lane & 15;
    const int rowBlk = blockIdx.x * 128;
    const int colBlk = blockIdx.y * 64;

    v8f acc[2][2];
    #pragma unroll
    for (int rg = 0; rg < 2; ++rg)
        #pragma unroll
        for (int cg = 0; cg < 2; ++cg)
            #pragma unroll
            for (int v = 0; v < 8; ++v) acc[rg][cg][v] = 0.f;

    for (int kt = 0; kt < K; kt += 32) {
        // A tile: 128x32, 16 bf16 per thread via two async b128
        {
            int ar = t >> 1, ac = (t & 1) * 16;
            const __bf16* src = A + (size_t)(rowBlk + ar) * K + kt + ac;
            async_ld_b128(&As[ar * 40 + ac], src);
            async_ld_b128(&As[ar * 40 + ac + 8], src + 8);
        }
        // B tile: 64x32
        if (t < 128) {
            int br = t >> 1, bc = (t & 1) * 16;
            const __bf16* src = Bw + (size_t)(colBlk + br) * K + kt + bc;
            async_ld_b128(&Bs[br * 40 + bc], src);
            async_ld_b128(&Bs[br * 40 + bc + 8], src + 8);
        }
        wait_async0();
        __syncthreads();

        BFrag af[2], bfr[2];
        #pragma unroll
        for (int rg = 0; rg < 2; ++rg) {
            const char* base = (const char*)&As[(wm * 32 + rg * 16 + llo) * 40];
            af[rg].q[0] = *(const uint4*)(base + lhi * 16);        // K = kb..kb+7
            af[rg].q[1] = *(const uint4*)(base + 32 + lhi * 16);   // K = 16+kb..
        }
        #pragma unroll
        for (int cg = 0; cg < 2; ++cg) {
            const char* base = (const char*)&Bs[(wn * 32 + cg * 16 + llo) * 40];
            bfr[cg].q[0] = *(const uint4*)(base + lhi * 16);
            bfr[cg].q[1] = *(const uint4*)(base + 32 + lhi * 16);
        }
        #pragma unroll
        for (int rg = 0; rg < 2; ++rg)
            #pragma unroll
            for (int cg = 0; cg < 2; ++cg)
                acc[rg][cg] = wmma_bf16(af[rg].v, bfr[cg].v, acc[rg][cg]);
        __syncthreads();
    }

    // Epilogue.  C layout: row = base + lhi*8 + v, col = base + llo.
    #pragma unroll
    for (int rg = 0; rg < 2; ++rg) {
        #pragma unroll
        for (int cg = 0; cg < 2; ++cg) {
            #pragma unroll
            for (int v = 0; v < 8; ++v) {
                int row = rowBlk + wm * 32 + rg * 16 + lhi * 8 + v;
                int col = colBlk + wn * 32 + cg * 16 + llo;
                float val = acc[rg][cg][v];
                if (EPI == 0) {
                    float other = __shfl_xor(val, 1, 32); // RoPE pair partner (col^1)
                    if (ropeFlag) {
                        int p = (col & 31) >> 1;
                        int s = row & (S_LEN - 1);
                        float invf = __expf(-0.57564627325f * (float)p); // 10000^(-p/16)
                        float sn, cs;
                        __sincosf((float)s * invf, &sn, &cs);
                        val = (col & 1) ? (val * cs + other * sn)
                                        : (val * cs - other * sn);
                    }
                    int b = row >> 12, s = row & (S_LEN - 1);
                    int h = col >> 5,  d = col & 31;
                    ((__bf16*)outp)[(((size_t)(b * NHEAD + h) * S_LEN + s) << 5) + d] = (__bf16)val;
                } else if (EPI == 1) {
                    size_t idx = (size_t)row * N + col;
                    ((float*)outp)[idx] = val + resid[idx];
                } else if (EPI == 2) {
                    float u = val + bias[col];
                    float gl = 0.5f * u * (1.f + erff(u * 0.70710678118f));
                    ((__bf16*)outp)[(size_t)row * N + col] = (__bf16)gl;
                } else { // EPI == 3
                    size_t idx = (size_t)row * N + col;
                    ((float*)outp)[idx] = val + bias[col] + resid[idx];
                }
            }
        }
    }
}

// ---------------------------------------------------------------------------
// Flash attention. Q,K,V bf16 [B,H,S,32]. 4 waves/block, 32 q-rows per wave.
// KV streamed in 32-row chunks. Softmax: wave32 row-ownership (lane L owns
// q-row L of the wave tile): score tile spilled to wave-private LDS in
// C-layout, owner lane reduces its row serially, emits bf16 P row + alpha.
// No cross-lane permutes. Same-wave LDS RAW is in-order (DScnt).
// ---------------------------------------------------------------------------
__global__ __launch_bounds__(128)
void attention(const __bf16* __restrict__ Qh, const __bf16* __restrict__ Kh,
               const __bf16* __restrict__ Vh, __bf16* __restrict__ Oout) {
    const int bh  = blockIdx.y;
    const int qt  = blockIdx.x;
    const int t   = threadIdx.x;
    const int w   = t >> 5;
    const int lane = t & 31;
    const int lhi = lane >> 4, llo = lane & 15;

    __shared__ __bf16 ldsK[32 * 40];          // K chunk, row-major
    __shared__ __bf16 ldsVt[32 * 40];         // V chunk, transposed [dh][kv]
    __shared__ float  ldsS[4][32 * 36];       // per-wave raw scores (f32)
    __shared__ __bf16 ldsP[4][32 * 40];       // per-wave P tile (bf16)
    __shared__ float  ldsA[4][32];            // per-wave alpha / 1/l broadcast

    const __bf16* Qb = Qh + (size_t)bh * S_LEN * DHEAD;
    const __bf16* Kb = Kh + (size_t)bh * S_LEN * DHEAD;
    const __bf16* Vb = Vh + (size_t)bh * S_LEN * DHEAD;

    const int qbase = qt * 128 + w * 32;
    const float scl = 0.17677669529663689f;   // 1/sqrt(32)

    // Q fragments (held in registers for the whole KV loop)
    BFrag aq[2];
    #pragma unroll
    for (int rg = 0; rg < 2; ++rg) {
        const char* qrow = (const char*)(Qb + (size_t)(qbase + rg * 16 + llo) * DHEAD);
        aq[rg].q[0] = *(const uint4*)(qrow + lhi * 16);
        aq[rg].q[1] = *(const uint4*)(qrow + 32 + lhi * 16);
    }

    v8f o[2][2];
    #pragma unroll
    for (int rg = 0; rg < 2; ++rg)
        #pragma unroll
        for (int cg = 0; cg < 2; ++cg)
            #pragma unroll
            for (int v = 0; v < 8; ++v) o[rg][cg][v] = 0.f;
    float mrun = -1e30f, lrun = 0.f;          // stats of q-row (qbase + lane)

    for (int kv = 0; kv < S_LEN; kv += 32) {
        // --- stage K (async copy-through) and V (register transpose) ---
        {
            int r = t >> 2, cb = (t & 3) * 8;
            async_ld_b128(&ldsK[r * 40 + cb], Kb + (size_t)(kv + r) * DHEAD + cb);
            union { uint4 q; __bf16 h[8]; } vv;
            vv.q = *(const uint4*)(Vb + (size_t)(kv + r) * DHEAD + cb);
            #pragma unroll
            for (int j = 0; j < 8; ++j) ldsVt[(cb + j) * 40 + r] = vv.h[j];
        }
        wait_async0();
        __syncthreads();

        // --- scores: S[32q x 32k] via 4 WMMA ---
        BFrag bk[2];
        #pragma unroll
        for (int cg = 0; cg < 2; ++cg) {
            const char* base = (const char*)&ldsK[(cg * 16 + llo) * 40];
            bk[cg].q[0] = *(const uint4*)(base + lhi * 16);
            bk[cg].q[1] = *(const uint4*)(base + 32 + lhi * 16);
        }
        v8f sc[2][2];
        #pragma unroll
        for (int rg = 0; rg < 2; ++rg)
            #pragma unroll
            for (int cg = 0; cg < 2; ++cg) {
                v8f z = {};
                sc[rg][cg] = wmma_bf16(aq[rg].v, bk[cg].v, z);
            }

        // --- spill scaled scores (C-layout) to wave-private LDS ---
        #pragma unroll
        for (int rg = 0; rg < 2; ++rg)
            #pragma unroll
            for (int cg = 0; cg < 2; ++cg)
                #pragma unroll
                for (int v = 0; v < 8; ++v)
                    ldsS[w][(rg * 16 + lhi * 8 + v) * 36 + cg * 16 + llo] = sc[rg][cg][v] * scl;
        lds_fence();

        // --- owner pass: lane L reduces q-row L (32 values) ---
        float sv[32];
        {
            const float4* srow = (const float4*)&ldsS[w][lane * 36];
            #pragma unroll
            for (int i = 0; i < 8; ++i) {
                float4 f4 = srow[i];
                sv[4 * i + 0] = f4.x; sv[4 * i + 1] = f4.y;
                sv[4 * i + 2] = f4.z; sv[4 * i + 3] = f4.w;
            }
        }
        float red[16];
        #pragma unroll
        for (int j = 0; j < 16; ++j) red[j] = fmaxf(sv[j], sv[j + 16]);
        #pragma unroll
        for (int st = 8; st >= 1; st >>= 1)
            #pragma unroll
            for (int j = 0; j < st; ++j) red[j] = fmaxf(red[j], red[j + st]);
        float mnew  = fmaxf(mrun, red[0]);
        float alpha = __expf(mrun - mnew);
        mrun = mnew;

        union { uint4 q[4]; __bf16 h[32]; } pk;
        float pv[32];
        #pragma unroll
        for (int j = 0; j < 32; ++j) {
            pv[j] = __expf(sv[j] - mnew);
            pk.h[j] = (__bf16)pv[j];
        }
        #pragma unroll
        for (int j = 0; j < 16; ++j) red[j] = pv[j] + pv[j + 16];
        #pragma unroll
        for (int st = 8; st >= 1; st >>= 1)
            #pragma unroll
            for (int j = 0; j < st; ++j) red[j] += red[j + st];
        lrun = lrun * alpha + red[0];

        {
            uint4* pdst = (uint4*)&ldsP[w][lane * 40];
            pdst[0] = pk.q[0]; pdst[1] = pk.q[1];
            pdst[2] = pk.q[2]; pdst[3] = pk.q[3];
        }
        ldsA[w][lane] = alpha;
        lds_fence();

        // --- rescale O by alpha of each row ---
        #pragma unroll
        for (int rg = 0; rg < 2; ++rg) {
            const float4* a4 = (const float4*)&ldsA[w][rg * 16 + lhi * 8];
            float4 a0 = a4[0], a1 = a4[1];
            float av[8] = {a0.x, a0.y, a0.z, a0.w, a1.x, a1.y, a1.z, a1.w};
            #pragma unroll
            for (int cg = 0; cg < 2; ++cg)
                #pragma unroll
                for (int v = 0; v < 8; ++v) o[rg][cg][v] *= av[v];
        }

        // --- P @ V ---
        BFrag bv[2];
        #pragma unroll
        for (int cgd = 0; cgd < 2; ++cgd) {
            const char* base = (const char*)&ldsVt[(cgd * 16 + llo) * 40];
            bv[cgd].q[0] = *(const uint4*)(base + lhi * 16);
            bv[cgd].q[1] = *(const uint4*)(base + 32 + lhi * 16);
        }
        #pragma unroll
        for (int rg = 0; rg < 2; ++rg) {
            BFrag ap;
            const char* base = (const char*)&ldsP[w][(rg * 16 + llo) * 40];
            ap.q[0] = *(const uint4*)(base + lhi * 16);
            ap.q[1] = *(const uint4*)(base + 32 + lhi * 16);
            #pragma unroll
            for (int cgd = 0; cgd < 2; ++cgd)
                o[rg][cgd] = wmma_bf16(ap.v, bv[cgd].v, o[rg][cgd]);
        }
        __syncthreads();
    }

    // --- finalize: broadcast 1/l via LDS, normalize, store [B*S, 256] ---
    ldsA[w][lane] = 1.0f / lrun;
    lds_fence();
    const int b = bh >> 3, h = bh & 7;
    #pragma unroll
    for (int rg = 0; rg < 2; ++rg) {
        const float4* a4 = (const float4*)&ldsA[w][rg * 16 + lhi * 8];
        float4 a0 = a4[0], a1 = a4[1];
        float iv[8] = {a0.x, a0.y, a0.z, a0.w, a1.x, a1.y, a1.z, a1.w};
        #pragma unroll
        for (int v = 0; v < 8; ++v) {
            int srow = qbase + rg * 16 + lhi * 8 + v;
            size_t orow = ((size_t)b * S_LEN + srow) * DMODEL;
            #pragma unroll
            for (int cgd = 0; cgd < 2; ++cgd)
                Oout[orow + h * 32 + cgd * 16 + llo] = (__bf16)(o[rg][cgd][v] * iv[v]);
        }
    }
}

// ---------------------------------------------------------------------------
// Host launcher
// ---------------------------------------------------------------------------
extern "C" void kernel_launch(void* const* d_in, const int* in_sizes, int n_in,
                              void* d_out, int out_size, void* d_ws, size_t ws_size,
                              hipStream_t stream) {
    const float* x    = (const float*)d_in[0];
    const float* Wq   = (const float*)d_in[1];
    const float* Wk   = (const float*)d_in[2];
    const float* Wv   = (const float*)d_in[3];
    const float* Wo   = (const float*)d_in[4];
    const float* ln1g = (const float*)d_in[5];
    const float* ln1b = (const float*)d_in[6];
    const float* ln2g = (const float*)d_in[7];
    const float* ln2b = (const float*)d_in[8];
    const float* W2   = (const float*)d_in[9];
    const float* b2   = (const float*)d_in[10];
    const float* W3   = (const float*)d_in[11];
    const float* b3   = (const float*)d_in[12];
    float* out = (float*)d_out;

    const int M  = 2 * S_LEN;
    const int D  = DMODEL;
    const int DM = 1024;

    char* ws = (char*)d_ws;
    size_t off = 0;
    auto alloc = [&](size_t bytes) -> void* {
        void* p = ws + off;
        off += (bytes + 255) & ~(size_t)255;
        return p;
    };
    __bf16* wqb  = (__bf16*)alloc((size_t)D * D * 2);
    __bf16* wkb  = (__bf16*)alloc((size_t)D * D * 2);
    __bf16* wvb  = (__bf16*)alloc((size_t)D * D * 2);
    __bf16* wob  = (__bf16*)alloc((size_t)D * D * 2);
    __bf16* w2b  = (__bf16*)alloc((size_t)DM * D * 2);
    __bf16* w3b  = (__bf16*)alloc((size_t)D * DM * 2);
    __bf16* xn   = (__bf16*)alloc((size_t)M * D * 2);
    __bf16* Qb   = (__bf16*)alloc((size_t)M * D * 2);   // [B,H,S,32]
    __bf16* Kb   = (__bf16*)alloc((size_t)M * D * 2);
    __bf16* Vb   = (__bf16*)alloc((size_t)M * D * 2);
    __bf16* attn = (__bf16*)alloc((size_t)M * D * 2);
    float*  mlpi = (float*) alloc((size_t)M * D * 4);
    __bf16* y2   = (__bf16*)alloc((size_t)M * D * 2);
    __bf16* hmid = (__bf16*)alloc((size_t)M * DM * 2);
    (void)ws_size; (void)in_sizes; (void)n_in; (void)out_size;

    cast_f32_bf16<<<(D * D + 255) / 256, 256, 0, stream>>>(Wq, wqb, D * D);
    cast_f32_bf16<<<(D * D + 255) / 256, 256, 0, stream>>>(Wk, wkb, D * D);
    cast_f32_bf16<<<(D * D + 255) / 256, 256, 0, stream>>>(Wv, wvb, D * D);
    cast_f32_bf16<<<(D * D + 255) / 256, 256, 0, stream>>>(Wo, wob, D * D);
    cast_f32_bf16<<<(DM * D + 255) / 256, 256, 0, stream>>>(W2, w2b, DM * D);
    cast_f32_bf16<<<(DM * D + 255) / 256, 256, 0, stream>>>(W3, w3b, DM * D);

    layernorm_bf16<<<M, 256, 0, stream>>>(x, ln1g, ln1b, xn);

    dim3 gP(M / 128, D / 64);
    gemm_nt<0><<<gP, 256, 0, stream>>>(xn, wqb, M, D, D, nullptr, nullptr, Qb, 1);
    gemm_nt<0><<<gP, 256, 0, stream>>>(xn, wkb, M, D, D, nullptr, nullptr, Kb, 1);
    gemm_nt<0><<<gP, 256, 0, stream>>>(xn, wvb, M, D, D, nullptr, nullptr, Vb, 0);

    attention<<<dim3(S_LEN / 128, 2 * NHEAD), 128, 0, stream>>>(Qb, Kb, Vb, attn);

    gemm_nt<1><<<gP, 256, 0, stream>>>(attn, wob, M, D, D, nullptr, x, mlpi, 0);

    layernorm_bf16<<<M, 256, 0, stream>>>(mlpi, ln2g, ln2b, y2);

    gemm_nt<2><<<dim3(M / 128, DM / 64), 256, 0, stream>>>(y2, w2b, M, DM, D, b2, nullptr, hmid, 0);

    gemm_nt<3><<<dim3(M / 128, D / 64), 256, 0, stream>>>(hmid, w3b, M, D, DM, b3, mlpi, out, 0);
}
